// GATGuard_39762807226542
// MI455X (gfx1250) — matrix operand order repaired and checked by reading, not compiled, not run on hardware
//
#include <hip/hip_runtime.h>
#include <hip/hip_bf16.h>

#define NN 50000
#define EE 800000

typedef __attribute__((ext_vector_type(16))) __bf16 v16bf;
typedef __attribute__((ext_vector_type(8)))  float  v8f;

union FragBF { v16bf v; uint4 q[2]; };

__device__ __forceinline__ float lrelu(float x, float s) { return x > 0.f ? x : x * s; }

__device__ __forceinline__ __bf16 f2bf(float f) {
    unsigned u = __float_as_uint(f);
    unsigned r = u + 0x7FFFu + ((u >> 16) & 1u);   // round-to-nearest-even
    unsigned short h = (unsigned short)(r >> 16);
    return __builtin_bit_cast(__bf16, h);
}

__device__ __forceinline__ void atomicMaxF(float* addr, float v) {
    if (v >= 0.f) atomicMax((int*)addr, __float_as_int(v));
    else          atomicMin((unsigned int*)addr, __float_as_uint(v));
}

// ---------------- mask / norm kernels ----------------

__global__ void mask_init_kernel(unsigned char* mask, int E) {
    int i = blockIdx.x * blockDim.x + threadIdx.x;
    if (i < E) mask[i] = 1;
}

__global__ __launch_bounds__(256) void rownorm_kernel(const float* __restrict__ X,
                                                      float* __restrict__ inv, int N, int F) {
    int wid  = (blockIdx.x * blockDim.x + threadIdx.x) >> 5;
    int lane = threadIdx.x & 31;
    if (wid >= N) return;
    const float* x = X + (size_t)wid * F;
    float s = 0.f;
    for (int k = lane; k < F; k += 32) { float v = x[k]; s += v * v; }
    for (int m = 16; m > 0; m >>= 1) s += __shfl_xor(s, m, 32);
    if (lane == 0) inv[wid] = rsqrtf(fmaxf(s, 1e-24f));
}

__global__ __launch_bounds__(256) void cos_mask_kernel(const float* __restrict__ X,
                                                       const float* __restrict__ inv,
                                                       const int* __restrict__ row,
                                                       const int* __restrict__ col,
                                                       unsigned char* __restrict__ mask,
                                                       int E, int F) {
    int wid  = (blockIdx.x * blockDim.x + threadIdx.x) >> 5;
    int lane = threadIdx.x & 31;
    if (wid >= E) return;
    if (!mask[wid]) return;
    int r = row[wid], c = col[wid];
    const float* xr = X + (size_t)r * F;
    const float* xc = X + (size_t)c * F;
    float s = 0.f;
    for (int k = lane; k < F; k += 32) s += xr[k] * xc[k];
    for (int m = 16; m > 0; m >>= 1) s += __shfl_xor(s, m, 32);
    if (lane == 0) {
        float sim = s * inv[r] * inv[c];
        if (sim < 0.1f) mask[wid] = 0;
    }
}

// ---------------- bf16 conversion ----------------

__global__ void to_bf16_kernel(const float* __restrict__ X, __bf16* __restrict__ Xb, size_t n) {
    size_t i = (size_t)blockIdx.x * blockDim.x + threadIdx.x;
    if (i < n) Xb[i] = f2bf(X[i]);
}

// W:[K,M] fp32 row-major -> Wt:[M,K] bf16 (so B fragments are contiguous in K)
__global__ void wtrans_bf16_kernel(const float* __restrict__ W, __bf16* __restrict__ Wt,
                                   int K, int M) {
    int i = blockIdx.x * blockDim.x + threadIdx.x;
    if (i >= K * M) return;
    int k = i / M, m = i % M;
    Wt[(size_t)m * K + k] = f2bf(W[i]);
}

// ---------------- WMMA GEMM: C[N,M] = A[N,K] * Wt^T ----------------
// One wave -> 16 rows x 64 cols (4 accumulators), K-step 32 (v_wmma_f32_16x16x32_bf16).
__global__ __launch_bounds__(256) void gemm_wmma_kernel(const __bf16* __restrict__ A,
                                                        const __bf16* __restrict__ Bt,
                                                        float* __restrict__ C,
                                                        int N, int K, int M) {
    int wave = threadIdx.x >> 5;
    int lane = threadIdx.x & 31;
    int lrow = lane & 15;
    int lhalf = lane >> 4;
    int rowBase = (blockIdx.x * 8 + wave) * 16;
    int colBase = blockIdx.y * 64;

    v8f acc0 = {0,0,0,0,0,0,0,0}, acc1 = acc0, acc2 = acc0, acc3 = acc0;

    int arow = rowBase + lrow;
    const __bf16* aptr = A + (size_t)(arow < N ? arow : 0) * K;
    const __bf16* b0p = Bt + (size_t)(colBase +  0 + lrow) * K + lhalf * 16;
    const __bf16* b1p = Bt + (size_t)(colBase + 16 + lrow) * K + lhalf * 16;
    const __bf16* b2p = Bt + (size_t)(colBase + 32 + lrow) * K + lhalf * 16;
    const __bf16* b3p = Bt + (size_t)(colBase + 48 + lrow) * K + lhalf * 16;

    for (int k0 = 0; k0 < K; k0 += 32) {
        FragBF a;
        // A 16x32 bf16 layout: lane half 0 -> K 0..15 contiguous; half 1 -> K 16..31
        a.q[0] = *(const uint4*)(aptr + k0 +     lhalf * 16);
        a.q[1] = *(const uint4*)(aptr + k0 + 8 + lhalf * 16);
        FragBF b;
        // B 32x16 bf16 layout: lane half 0 -> K 0..15 contiguous; half 1 -> K 16..31
        b.q[0] = *(const uint4*)(b0p + k0); b.q[1] = *(const uint4*)(b0p + k0 + 8);
        acc0 = __builtin_amdgcn_wmma_f32_16x16x32_bf16(false, a.v, false, b.v, (short)0, acc0, false, false);
        b.q[0] = *(const uint4*)(b1p + k0); b.q[1] = *(const uint4*)(b1p + k0 + 8);
        acc1 = __builtin_amdgcn_wmma_f32_16x16x32_bf16(false, a.v, false, b.v, (short)0, acc1, false, false);
        b.q[0] = *(const uint4*)(b2p + k0); b.q[1] = *(const uint4*)(b2p + k0 + 8);
        acc2 = __builtin_amdgcn_wmma_f32_16x16x32_bf16(false, a.v, false, b.v, (short)0, acc2, false, false);
        b.q[0] = *(const uint4*)(b3p + k0); b.q[1] = *(const uint4*)(b3p + k0 + 8);
        acc3 = __builtin_amdgcn_wmma_f32_16x16x32_bf16(false, a.v, false, b.v, (short)0, acc3, false, false);
    }

    // D layout: VGPR v, lane half g -> row = rowBase + 8*g + v, col = colBase + t*16 + lrow
    v8f accs[4] = {acc0, acc1, acc2, acc3};
    #pragma unroll
    for (int t = 0; t < 4; ++t) {
        int col = colBase + t * 16 + lrow;
        #pragma unroll
        for (int v = 0; v < 8; ++v) {
            int r = rowBase + 8 * lhalf + v;
            if (r < N) C[(size_t)r * M + col] = accs[t][v];
        }
    }
}

// ---------------- attention scores ----------------

__global__ __launch_bounds__(256) void score_kernel(const float* __restrict__ feat,
                                                    const float* __restrict__ al,
                                                    const float* __restrict__ ar,
                                                    float* __restrict__ el, float* __restrict__ er,
                                                    float* __restrict__ eself, float* __restrict__ mmax,
                                                    int N, int H, int D) {
    int wid  = (blockIdx.x * blockDim.x + threadIdx.x) >> 5;
    int lane = threadIdx.x & 31;
    if (wid >= N) return;
    const float* f = feat + (size_t)wid * H * D;
    for (int h = 0; h < H; ++h) {
        float sl = 0.f, sr = 0.f;
        for (int d = lane; d < D; d += 32) {
            float v = f[h * D + d];
            sl += v * al[h * D + d];
            sr += v * ar[h * D + d];
        }
        for (int m = 16; m > 0; m >>= 1) { sl += __shfl_xor(sl, m, 32); sr += __shfl_xor(sr, m, 32); }
        if (lane == 0) {
            el[wid * H + h] = sl;
            er[wid * H + h] = sr;
            float es = lrelu(sl + sr, 0.2f);
            eself[wid * H + h] = es;
            mmax[wid * H + h]  = es;   // segment max init = self-loop score
        }
    }
}

__global__ void edge_max_kernel(const int* __restrict__ row, const int* __restrict__ col,
                                const unsigned char* __restrict__ mask,
                                const float* __restrict__ el, const float* __restrict__ er,
                                float* __restrict__ mmax, int E, int H) {
    int e = blockIdx.x * blockDim.x + threadIdx.x;
    if (e >= E || !mask[e]) return;
    int r = row[e], c = col[e];
    for (int h = 0; h < H; ++h) {
        float sc = lrelu(el[r * H + h] + er[c * H + h], 0.2f);
        atomicMaxF(&mmax[c * H + h], sc);
    }
}

__global__ void denom_init_kernel(const float* __restrict__ eself, const float* __restrict__ mmax,
                                  float* __restrict__ denom, int NH) {
    int i = blockIdx.x * blockDim.x + threadIdx.x;
    if (i < NH) denom[i] = __expf(eself[i] - mmax[i]);
}

__global__ void edge_denom_kernel(const int* __restrict__ row, const int* __restrict__ col,
                                  const unsigned char* __restrict__ mask,
                                  const float* __restrict__ el, const float* __restrict__ er,
                                  const float* __restrict__ mmax, float* __restrict__ denom,
                                  int E, int H) {
    int e = blockIdx.x * blockDim.x + threadIdx.x;
    if (e >= E || !mask[e]) return;
    int r = row[e], c = col[e];
    for (int h = 0; h < H; ++h) {
        float sc = lrelu(el[r * H + h] + er[c * H + h], 0.2f);
        atomicAdd(&denom[c * H + h], __expf(sc - mmax[c * H + h]));
    }
}

// self-loop term + bias: rst[n,j] = (exp_s/denom)*feat[n,j] + b[j]
__global__ void rst_init_kernel(const float* __restrict__ feat, const float* __restrict__ eself,
                                const float* __restrict__ mmax, const float* __restrict__ denom,
                                const float* __restrict__ bias, float* __restrict__ rst,
                                int N, int HD, int H) {
    size_t i = (size_t)blockIdx.x * blockDim.x + threadIdx.x;
    if (i >= (size_t)N * HD) return;
    int n = (int)(i / HD), j = (int)(i % HD);
    int h = j >> 6;                       // D == 64 in all layers
    float coef = __expf(eself[n * H + h] - mmax[n * H + h]) / denom[n * H + h];
    rst[i] = coef * feat[i] + bias[j];
}

__global__ __launch_bounds__(256) void edge_agg_kernel(const int* __restrict__ row,
                                                       const int* __restrict__ col,
                                                       const unsigned char* __restrict__ mask,
                                                       const float* __restrict__ el,
                                                       const float* __restrict__ er,
                                                       const float* __restrict__ mmax,
                                                       const float* __restrict__ denom,
                                                       const float* __restrict__ feat,
                                                       float* __restrict__ rst,
                                                       int E, int H, int HD) {
    int wid  = (blockIdx.x * blockDim.x + threadIdx.x) >> 5;
    int lane = threadIdx.x & 31;
    if (wid >= E) return;
    if (!mask[wid]) return;
    int r = row[wid], c = col[wid];
    const float* fr = feat + (size_t)r * HD;
    float*       rc = rst  + (size_t)c * HD;
    for (int j = lane; j < HD; j += 32) {
        int h = j >> 6;                   // D == 64
        float sc = lrelu(el[r * H + h] + er[c * H + h], 0.2f);
        float w  = __expf(sc - mmax[c * H + h]) / denom[c * H + h];
        atomicAdd(&rc[j], w * fr[j]);
    }
}

__global__ void act_kernel(const float* __restrict__ rst, float* __restrict__ xout, size_t n) {
    size_t i = (size_t)blockIdx.x * blockDim.x + threadIdx.x;
    if (i < n) xout[i] = lrelu(rst[i], 0.01f);
}

// ---------------- host launch ----------------

extern "C" void kernel_launch(void* const* d_in, const int* in_sizes, int n_in,
                              void* d_out, int out_size, void* d_ws, size_t ws_size,
                              hipStream_t stream) {
    const int N = NN, E = EE, F0 = 128, HD = 256, C = 64;

    const float* x0  = (const float*)d_in[0];
    const int* erow  = (const int*)d_in[1];
    const int* ecol  = (const int*)d_in[2];
    const float* Ws[3]  = { (const float*)d_in[3], (const float*)d_in[7],  (const float*)d_in[11] };
    const float* als[3] = { (const float*)d_in[4], (const float*)d_in[8],  (const float*)d_in[12] };
    const float* ars[3] = { (const float*)d_in[5], (const float*)d_in[9],  (const float*)d_in[13] };
    const float* bs[3]  = { (const float*)d_in[6], (const float*)d_in[10], (const float*)d_in[14] };
    const int Fin[3]  = { F0, HD, HD };
    const int Kout[3] = { HD, HD, C };
    const int Hs[3]   = { 4, 4, 1 };

    char* ws = (char*)d_ws;
    size_t off = 0;
    auto alloc = [&](size_t bytes) -> void* {
        off = (off + 255) & ~(size_t)255;
        void* p = ws + off;
        off += bytes;
        return p;
    };
    float*  xA    = (float*)alloc((size_t)N * HD * 4);
    float*  xB    = (float*)alloc((size_t)N * HD * 4);
    float*  feat  = (float*)alloc((size_t)N * HD * 4);
    float*  rstT  = (float*)alloc((size_t)N * HD * 4);
    __bf16* xbf   = (__bf16*)alloc((size_t)N * HD * 2);
    __bf16* Wt    = (__bf16*)alloc((size_t)HD * HD * 2);
    float*  inv   = (float*)alloc((size_t)N * 4);
    float*  el    = (float*)alloc((size_t)N * 4 * 4);
    float*  er    = (float*)alloc((size_t)N * 4 * 4);
    float*  esf   = (float*)alloc((size_t)N * 4 * 4);
    float*  mmax  = (float*)alloc((size_t)N * 4 * 4);
    float*  denom = (float*)alloc((size_t)N * 4 * 4);
    unsigned char* mask = (unsigned char*)alloc((size_t)E);

    const float* xin[3]  = { x0, xA, xB };
    float*       rst[3]  = { rstT, rstT, (float*)d_out };
    float*       xout[3] = { xA, xB, nullptr };

    mask_init_kernel<<<(E + 255) / 256, 256, 0, stream>>>(mask, E);

    for (int L = 0; L < 3; ++L) {
        int Fi = Fin[L], Ko = Kout[L], H = Hs[L];
        // 1) cosine edge mask (accumulative)
        rownorm_kernel<<<(N * 32 + 255) / 256, 256, 0, stream>>>(xin[L], inv, N, Fi);
        cos_mask_kernel<<<(E * 32 + 255) / 256, 256, 0, stream>>>(xin[L], inv, erow, ecol, mask, E, Fi);
        // 2) bf16 staging for WMMA
        size_t nx = (size_t)N * Fi;
        to_bf16_kernel<<<(unsigned)((nx + 255) / 256), 256, 0, stream>>>(xin[L], xbf, nx);
        wtrans_bf16_kernel<<<(Fi * Ko + 255) / 256, 256, 0, stream>>>(Ws[L], Wt, Fi, Ko);
        // 3) feat = x @ W via v_wmma_f32_16x16x32_bf16
        dim3 ggrid((N + 127) / 128, Ko / 64);
        gemm_wmma_kernel<<<ggrid, 256, 0, stream>>>(xbf, Wt, feat, N, Fi, Ko);
        // 4) per-node scores + segment-max init
        score_kernel<<<(N * 32 + 255) / 256, 256, 0, stream>>>(feat, als[L], ars[L], el, er, esf, mmax, N, H, 64);
        // 5) softmax over incoming edges (+ self loop)
        edge_max_kernel<<<(E + 255) / 256, 256, 0, stream>>>(erow, ecol, mask, el, er, mmax, E, H);
        denom_init_kernel<<<(N * H + 255) / 256, 256, 0, stream>>>(esf, mmax, denom, N * H);
        edge_denom_kernel<<<(E + 255) / 256, 256, 0, stream>>>(erow, ecol, mask, el, er, mmax, denom, E, H);
        // 6) aggregate
        rst_init_kernel<<<(unsigned)(((size_t)N * Ko + 255) / 256), 256, 0, stream>>>(
            feat, esf, mmax, denom, bs[L], rst[L], N, Ko, H);
        edge_agg_kernel<<<(E * 32 + 255) / 256, 256, 0, stream>>>(
            erow, ecol, mask, el, er, mmax, denom, feat, rst[L], E, H, Ko);
        // 7) activation into next-layer input (final layer writes d_out directly, no act)
        if (L < 2) {
            size_t no = (size_t)N * Ko;
            act_kernel<<<(unsigned)((no + 255) / 256), 256, 0, stream>>>(rst[L], xout[L], no);
        }
    }
    (void)in_sizes; (void)n_in; (void)out_size; (void)ws_size;
}
